// GateModMLP_60224031425146
// MI455X (gfx1250) — matrix operand order repaired and checked by reading, not compiled, not run on hardware
//
#include <hip/hip_runtime.h>

typedef __attribute__((ext_vector_type(16))) __bf16 v16bf;
typedef __attribute__((ext_vector_type(8)))  float  v8f;
typedef __attribute__((ext_vector_type(4)))  float  v4f;
typedef __attribute__((ext_vector_type(4)))  unsigned int v4u;
typedef __attribute__((ext_vector_type(8)))  int    v8i;
typedef __attribute__((ext_vector_type(4)))  int    v4i;
typedef unsigned short u16;
typedef unsigned int   u32;
typedef unsigned long long u64;

#define B_  10240
#define D_  512
#define M_  8
#define L_  3

// LDS activation block: 128 rows x 512 bf16, padded row stride for banking.
// Row = 256 dwords; TDM pad: interval=256 dwords (enc 7), amount=4 dwords (enc 3)
// => row stride 260 dwords = 520 halfs = 1040 B; stride mod 64 banks = 4, so the
// 16 rows of a WMMA A-fragment start on 16 distinct bank quads (b128-aligned).
#define ROWH_  520
#define LDS_BYTES_ (128 * 260 * 4)   // 133120

__device__ inline u16 f2bf(float f) {
  u32 u = __float_as_uint(f);
  u += 0x7FFFu + ((u >> 16) & 1u);
  return (u16)(u >> 16);
}

// ---------------------------------------------------------------------------
// Pack Ws (L,M,D,D) f32 -> bf16, pre-swizzled into WMMA B-fragment lane order.
// Fragment (m, s=ktile, nt=ntile): 512 halfs; entry (lane, q):
//   value = Ws[l][m][ i = nt*16 + lane%16 ][ j = s*32 + (lane/16)*16 + q ]
// so each lane reads its 16 halfs (32B) contiguously in the GEMM kernel.
// ---------------------------------------------------------------------------
__global__ __launch_bounds__(256) void pack_w_kernel(const float* __restrict__ Ws,
                                                     u16* __restrict__ wp) {
  int id   = blockIdx.x * 256 + threadIdx.x;   // over L*M*16*32*512 = L*M*2^18
  int q    = id & 15;
  int lane = (id >> 4) & 31;
  int nt   = (id >> 9) & 31;
  int s    = (id >> 14) & 15;
  int lm   = id >> 18;                         // l*M + m
  int n = nt * 16 + (lane & 15);
  int j = s * 32 + (lane >> 4) * 16 + q;
  wp[id] = f2bf(Ws[((size_t)lm * 512 + n) * 512 + j]);
}

// pWs (L,B,M) -> pwt (L,M,B) for contiguous per-row-tile vector loads
__global__ __launch_bounds__(256) void pack_pw_kernel(const float* __restrict__ pWs,
                                                      float* __restrict__ pwt) {
  int id = blockIdx.x * 256 + threadIdx.x;     // over L*M*B, dst [l][m][b]
  int b  = id % B_;
  int lm = id / B_;
  int l  = lm >> 3, m = lm & 7;
  pwt[id] = pWs[((size_t)l * B_ + b) * M_ + m];
}

// x f32 -> bf16 row-major
__global__ __launch_bounds__(256) void pack_x_kernel(const float* __restrict__ x,
                                                     u16* __restrict__ xp) {
  int id = blockIdx.x * 256 + threadIdx.x;
  xp[id] = f2bf(x[id]);
}

// biasbuf[b,i] = gbs[b,i] * sum_m pbs[b,m] * bs[m,i]
__global__ __launch_bounds__(256) void bias_kernel(const float* __restrict__ pbs_l,
                                                   const float* __restrict__ bs_l,
                                                   const float* __restrict__ gbs_l,
                                                   float* __restrict__ biasbuf) {
  int id  = blockIdx.x * 256 + threadIdx.x;
  int row = id >> 9, col = id & 511;
  float s = 0.f;
#pragma unroll
  for (int m = 0; m < M_; ++m) s = fmaf(pbs_l[row * M_ + m], bs_l[m * D_ + col], s);
  biasbuf[id] = gbs_l[id] * s;
}

// ---------------------------------------------------------------------------
// Main GEMM: out[b,i] = gw[b,i] * (sum_m pW[b,m] * sum_j Ws[m,i,j] x[b,j]) + bias[b,i]
// Tile: 128 rows x 64 cols per workgroup, 8 waves (2x2 16x16 tiles each),
// K-loop: m(8) x s(16 steps of 32), bf16 WMMA with f32 accum, per-m fold by pW.
// Activation block staged in LDS by the Tensor Data Mover (wave 0 issues the
// DMA; TENSORcnt + workgroup barrier synchronize), with TDM row padding to
// avoid LDS bank conflicts on the ds_load_b128 A-fragment reads.
// ---------------------------------------------------------------------------
__global__ __launch_bounds__(256) void gemm_kernel(const u16* __restrict__ xp,
                                                   const u16* __restrict__ wp_l,
                                                   const float* __restrict__ pwt_l,
                                                   const float* __restrict__ gw_l,
                                                   const float* __restrict__ biasbuf,
                                                   float* __restrict__ out) {
  extern __shared__ __align__(16) u16 smem[];
  const int tid    = threadIdx.x;
  const int rowBlk = blockIdx.x * 128;
  const int colBlk = blockIdx.y * 64;

#if __has_builtin(__builtin_amdgcn_tensor_load_to_lds) && \
    __has_builtin(__builtin_amdgcn_s_wait_tensorcnt)
  // ---- Tensor Data Mover staging: global (128x512 bf16, stride 512) -> LDS
  if ((tid >> 5) == 0) {                       // one wave issues the DMA
    const u64 ga  = (u64)(const void*)(xp + (size_t)rowBlk * D_);
    const u32 lds = __builtin_amdgcn_groupstaticsize();  // dynamic LDS base

    v4u g0;
    g0[0] = 1u;                                // count=1, user descriptor
    g0[1] = lds;                               // lds_addr
    g0[2] = (u32)ga;                           // global_addr[31:0]
    g0[3] = (u32)((ga >> 32) & 0x1FFFFFFu) | (2u << 30);  // addr[56:32] | type=2

    v8i g1;
    // data_size=2B (1<<16) | pad_enable (1<<20) | pad_interval=256dw (7<<22)
    // | pad_amount=4dw (3<<25); workgroup_mask=0 (not in cluster)
    g1[0] = (int)((1u << 16) | (1u << 20) | (7u << 22) | (3u << 25));
    g1[1] = (int)(512u << 16);                 // tensor_dim0 = 512   [79:48]
    g1[2] = (int)(128u << 16);                 // tensor_dim1 = 128   [111:80]
    g1[3] = (int)(512u << 16);                 // tile_dim0   = 512   [127:112]
    g1[4] = (int)(128u);                       // tile_dim1   = 128   [143:128]
    g1[5] = (int)(512u);                       // tensor_dim0_stride  [207:160]
    g1[6] = 0;
    g1[7] = 0;

    v4i gz4 = {0, 0, 0, 0};                    // groups 2/3 unused (2-D tile)
    v8i gz8 = {0, 0, 0, 0, 0, 0, 0, 0};
    __builtin_amdgcn_tensor_load_to_lds(g0, g1, gz4, gz4, gz8, 0);
    __builtin_amdgcn_s_wait_tensorcnt(0);
  }
#else
  // ---- Fallback: manual coalesced copy into the same padded layout
  {
    const uint4* src = (const uint4*)(xp + (size_t)rowBlk * D_);
    uint4* dst = (uint4*)smem;
#pragma unroll
    for (int i = 0; i < 32; ++i) {
      int id = i * 256 + tid;                  // 8192 uint4 = 128KB payload
      int r  = id >> 6, c8 = id & 63;          // row, 16B-chunk in row
      dst[r * 65 + c8] = src[r * 64 + c8];     // 65 uint4 = padded row stride
    }
  }
#endif
  __syncthreads();

  const int w = tid >> 5, lane = tid & 31;
  const int lh = lane >> 4, lr = lane & 15;
  const int wr = w >> 1, wc = w & 1;            // 4 row-groups x 2 col-groups

  union AF { v16bf v; uint4 u[2]; };
  union PW { v8f v; v4f f[2]; };

  const v8f vzero = {0.f, 0.f, 0.f, 0.f, 0.f, 0.f, 0.f, 0.f};
  v8f tot[2][2];
  tot[0][0] = vzero; tot[0][1] = vzero; tot[1][0] = vzero; tot[1][1] = vzero;

  const uint4* lds4 = (const uint4*)smem;
  // A-fragment per-lane LDS half-indices (padded row stride ROWH_)
  const int arow0 = (wr * 32 + lr) * ROWH_ + lh * 16;      // row-tile 0
  const int arow1 = arow0 + 16 * ROWH_;                    // row-tile 1
  const int ntg0  = blockIdx.y * 4 + wc * 2;               // global n-tile

  for (int m = 0; m < M_; ++m) {
    v8f part[2][2];
    part[0][0] = vzero; part[0][1] = vzero; part[1][0] = vzero; part[1][1] = vzero;

#pragma unroll 4
    for (int s = 0; s < 16; ++s) {
      AF a0, a1, b0, b1;
      const int ai0 = arow0 + s * 32;
      const int ai1 = arow1 + s * 32;
      a0.u[0] = lds4[ai0 >> 3];
      a0.u[1] = lds4[(ai0 + 8) >> 3];
      a1.u[0] = lds4[ai1 >> 3];
      a1.u[1] = lds4[(ai1 + 8) >> 3];

      const uint4* bp0 = (const uint4*)(wp_l + (((m * 16 + s) * 32 + ntg0) * 512 + lane * 16));
      const uint4* bp1 = (const uint4*)(wp_l + (((m * 16 + s) * 32 + ntg0 + 1) * 512 + lane * 16));
      b0.u[0] = bp0[0]; b0.u[1] = bp0[1];
      b1.u[0] = bp1[0]; b1.u[1] = bp1[1];

      part[0][0] = __builtin_amdgcn_wmma_f32_16x16x32_bf16(false, a0.v, false, b0.v,
                                                           (short)0, part[0][0], false, false);
      part[0][1] = __builtin_amdgcn_wmma_f32_16x16x32_bf16(false, a0.v, false, b1.v,
                                                           (short)0, part[0][1], false, false);
      part[1][0] = __builtin_amdgcn_wmma_f32_16x16x32_bf16(false, a1.v, false, b0.v,
                                                           (short)0, part[1][0], false, false);
      part[1][1] = __builtin_amdgcn_wmma_f32_16x16x32_bf16(false, a1.v, false, b1.v,
                                                           (short)0, part[1][1], false, false);
    }

    // Fold partials: tot[rt][ct][v] += pW[row(v), m] * part[rt][ct][v]
    // C layout: VGPR v holds row (tileRow + v + 8*lh)
    PW pw0, pw1;
    const float* pwbase = pwt_l + (size_t)m * B_ + rowBlk + wr * 32 + lh * 8;
    pw0.f[0] = ((const v4f*)pwbase)[0];
    pw0.f[1] = ((const v4f*)pwbase)[1];
    pw1.f[0] = ((const v4f*)(pwbase + 16))[0];
    pw1.f[1] = ((const v4f*)(pwbase + 16))[1];

    tot[0][0] += pw0.v * part[0][0];
    tot[0][1] += pw0.v * part[0][1];
    tot[1][0] += pw1.v * part[1][0];
    tot[1][1] += pw1.v * part[1][1];
  }

  // Epilogue: gate + bias
#pragma unroll
  for (int rtl = 0; rtl < 2; ++rtl) {
#pragma unroll
    for (int ctl = 0; ctl < 2; ++ctl) {
#pragma unroll
      for (int v = 0; v < 8; ++v) {
        const int row = rowBlk + wr * 32 + rtl * 16 + lh * 8 + v;
        const int col = colBlk + wc * 32 + ctl * 16 + lr;
        const int idx = row * D_ + col;
        out[idx] = fmaf(gw_l[idx], tot[rtl][ctl][v], biasbuf[idx]);
      }
    }
  }
}

// ---------------------------------------------------------------------------
// LayerNorm (no affine) + ELU over D=512, write next activation as bf16
// One block per row; 256 threads x 2 elements.
// ---------------------------------------------------------------------------
__global__ __launch_bounds__(256) void ln_elu_pack_kernel(const float* __restrict__ h,
                                                          u16* __restrict__ xp) {
  __shared__ float s1[256];
  __shared__ float s2[256];
  const int row = blockIdx.x, tid = threadIdx.x;
  const float a = h[row * D_ + tid];
  const float b = h[row * D_ + 256 + tid];
  s1[tid] = a + b;
  s2[tid] = a * a + b * b;
  __syncthreads();
#pragma unroll
  for (int off = 128; off >= 1; off >>= 1) {
    if (tid < off) { s1[tid] += s1[tid + off]; s2[tid] += s2[tid + off]; }
    __syncthreads();
  }
  const float mu   = s1[0] * (1.f / 512.f);
  const float var  = s2[0] * (1.f / 512.f) - mu * mu;
  const float rstd = rsqrtf(var + 1e-5f);
  float va = (a - mu) * rstd;
  float vb = (b - mu) * rstd;
  va = va > 0.f ? va : (__expf(va) - 1.f);
  vb = vb > 0.f ? vb : (__expf(vb) - 1.f);
  xp[row * D_ + tid]       = f2bf(va);
  xp[row * D_ + 256 + tid] = f2bf(vb);
}

// ---------------------------------------------------------------------------
extern "C" void kernel_launch(void* const* d_in, const int* in_sizes, int n_in,
                              void* d_out, int out_size, void* d_ws, size_t ws_size,
                              hipStream_t stream) {
  const float* x   = (const float*)d_in[0];
  const float* Ws  = (const float*)d_in[1];
  const float* bs  = (const float*)d_in[2];
  const float* pWs = (const float*)d_in[3];
  const float* pbs = (const float*)d_in[4];
  const float* gWs = (const float*)d_in[5];
  const float* gbs = (const float*)d_in[6];
  float* out = (float*)d_out;

  char* ws = (char*)d_ws;
  u16*   wp   = (u16*)ws;   ws += (size_t)L_ * M_ * D_ * D_ * 2;   // 12.6 MB
  float* pwt  = (float*)ws; ws += (size_t)L_ * M_ * B_ * 4;        // 1.0 MB
  u16*   xp   = (u16*)ws;   ws += (size_t)B_ * D_ * 2;             // 10.5 MB
  float* hbuf = (float*)ws; ws += (size_t)B_ * D_ * 4;             // 21 MB
  float* bias = (float*)ws; ws += (size_t)B_ * D_ * 4;             // 21 MB

  pack_w_kernel <<<L_ * M_ * 1024,       256, 0, stream>>>(Ws, wp);
  pack_pw_kernel<<<(L_ * M_ * B_) / 256, 256, 0, stream>>>(pWs, pwt);
  pack_x_kernel <<<(B_ * D_) / 256,      256, 0, stream>>>(x, xp);

  for (int l = 0; l < L_; ++l) {
    bias_kernel<<<(B_ * D_) / 256, 256, 0, stream>>>(
        pbs + (size_t)l * B_ * M_, bs + (size_t)l * M_ * D_,
        gbs + (size_t)l * B_ * D_, bias);

    float* o = (l == L_ - 1) ? out : hbuf;
    gemm_kernel<<<dim3(B_ / 128, D_ / 64), 256, LDS_BYTES_, stream>>>(
        xp, wp + (size_t)l * M_ * D_ * D_, pwt + (size_t)l * M_ * B_,
        gWs + (size_t)l * B_ * D_, bias, o);

    if (l != L_ - 1)
      ln_elu_pack_kernel<<<B_, 256, 0, stream>>>(hbuf, xp);
  }
}
